// GlobalMambaBlock_40346922778711
// MI455X (gfx1250) — matrix operand order repaired and checked by reading, not compiled, not run on hardware
//
#include <hip/hip_runtime.h>
#include <hip/hip_bf16.h>
#include <stdint.h>

// ---------------- problem constants ----------------
#define B_      2
#define L_      2048
#define C_      768
#define HW_     64          // H*W
#define DIN_    1536        // D_INNER
#define DST_    16          // D_STATE
#define DCONV_  4
#define DTR_    48          // DT_RANK
#define XDBW_   80          // DT_RANK + 2*D_STATE
#define BL_     (B_ * L_)   // 4096 rows everywhere

typedef _Float16 half_t;
typedef __attribute__((ext_vector_type(16))) half_t v16h;
typedef __attribute__((ext_vector_type(8)))  float  v8f;

union Frag { v16h h; uint4 q[2]; };

// ======================================================================
// WMMA GEMM:  Out[m][n] = sum_k A[m][k] * W[n][k]
// A: (M x K) f16 row-major, lda.  W: (N x K) f16 row-major, ldw.
// K % 64 == 0, M % 128 == 0.  Each wave -> 16 rows x (16*JT) cols.
// Branch-free inner loop, ping-pong software pipeline (2 K-steps/iter).
// Block = 256 threads = 8 waves. grid.y = M/128, grid.x = N-tiles.
// ======================================================================
template<int JT>
__global__ __launch_bounds__(256)
void wmma_gemm_kernel(const half_t* __restrict__ A, int lda,
                      const half_t* __restrict__ W, int ldw,
                      float* __restrict__ Out, int ldo,
                      int K, int n0base)
{
    const int wave = threadIdx.x >> 5;
    const int lane = threadIdx.x & 31;
    const int m0 = (blockIdx.y * 8 + wave) * 16;
    const int n0 = n0base + blockIdx.x * (16 * JT);

    const int row = lane & 15;           // A row / B col / C col within tile
    const int kg  = lane >> 4;           // K half-group select

    // A fragment: lanes 0-15 rows; q[0] = K[kg*8 .. +7], q[1] = K[16+kg*8 .. +7]
    const half_t* Ap = A + (size_t)(m0 + row) * lda + kg * 8;
    // B fragment: lane col = row; q[0] = K[kg*8 .. +7], q[1] = K[16+kg*8 .. +7]
    const half_t* Wp[JT];
    #pragma unroll
    for (int j = 0; j < JT; ++j)
        Wp[j] = W + (size_t)(n0 + 16 * j + row) * ldw + kg * 8;

    const v8f vzero = {};
    v8f acc[JT];
    #pragma unroll
    for (int j = 0; j < JT; ++j) acc[j] = vzero;

    Frag a0, a1;
    Frag b0[JT], b1[JT];

    // preload K-step 0 into buffer 0
    a0.q[0] = *(const uint4*)(Ap);
    a0.q[1] = *(const uint4*)(Ap + 16);
    #pragma unroll
    for (int j = 0; j < JT; ++j) {
        b0[j].q[0] = *(const uint4*)(Wp[j]);
        b0[j].q[1] = *(const uint4*)(Wp[j] + 16);
    }

    for (int k0 = 0; k0 < K; k0 += 64) {
        // issue loads for K-step k0+32 (always valid: K % 64 == 0)
        {
            const half_t* ap = Ap + k0 + 32;
            a1.q[0] = *(const uint4*)(ap);
            a1.q[1] = *(const uint4*)(ap + 16);
            #pragma unroll
            for (int j = 0; j < JT; ++j) {
                const half_t* wp = Wp[j] + k0 + 32;
                b1[j].q[0] = *(const uint4*)(wp);
                b1[j].q[1] = *(const uint4*)(wp + 16);
            }
        }
        #pragma unroll
        for (int j = 0; j < JT; ++j)
            acc[j] = __builtin_amdgcn_wmma_f32_16x16x32_f16(
                        false, a0.h, false, b0[j].h, (short)0, acc[j], false, false);

        if (k0 + 64 < K) {               // uniform; false only on last iteration
            const half_t* ap = Ap + k0 + 64;
            a0.q[0] = *(const uint4*)(ap);
            a0.q[1] = *(const uint4*)(ap + 16);
            #pragma unroll
            for (int j = 0; j < JT; ++j) {
                const half_t* wp = Wp[j] + k0 + 64;
                b0[j].q[0] = *(const uint4*)(wp);
                b0[j].q[1] = *(const uint4*)(wp + 16);
            }
        }
        #pragma unroll
        for (int j = 0; j < JT; ++j)
            acc[j] = __builtin_amdgcn_wmma_f32_16x16x32_f16(
                        false, a1.h, false, b1[j].h, (short)0, acc[j], false, false);
    }

    // C/D layout: VGPR r -> M = m0 + r + 8*kg, col = n0 + 16*j + (lane&15)
    #pragma unroll
    for (int j = 0; j < JT; ++j) {
        const int n = n0 + 16 * j + row;
        #pragma unroll
        for (int r = 0; r < 8; ++r)
            Out[(size_t)(m0 + r + 8 * kg) * ldo + n] = acc[j][r];
    }
}

static void launch_gemm(const half_t* A, int lda, const half_t* W, int ldw,
                        float* Out, int ldo, int M, int N, int K, hipStream_t s)
{
    const int nfull = N / 64;
    if (nfull > 0)
        wmma_gemm_kernel<4><<<dim3(nfull, M / 128), 256, 0, s>>>(
            A, lda, W, ldw, Out, ldo, K, 0);
    const int rem = N - nfull * 64;      // multiple of 16
    if (rem > 0)
        wmma_gemm_kernel<1><<<dim3(rem / 16, M / 128), 256, 0, s>>>(
            A, lda, W, ldw, Out, ldo, K, nfull * 64);
}

// ======================================================================
// xc = mean(x over H*W), emitted as f16. One wave per (b,l,c) element:
// 32 lanes * float2 = 64 contiguous floats, butterfly reduce.
// Streams 805 MB from HBM -> prefetch 32 KB ahead.
// ======================================================================
__global__ __launch_bounds__(256)
void mean_hw_kernel(const float* __restrict__ x, half_t* __restrict__ xc, long total)
{
    const long w = (long)blockIdx.x * 8 + (threadIdx.x >> 5);
    if (w >= total) return;
    const int lane = threadIdx.x & 31;
    __builtin_prefetch(x + (w + 128) * (long)HW_ + lane * 2, 0, 0);
    const float2 v = ((const float2*)(x + w * (long)HW_))[lane];
    float s = v.x + v.y;
    #pragma unroll
    for (int off = 16; off > 0; off >>= 1) s += __shfl_xor(s, off, 32);
    if (lane == 0) xc[w] = (half_t)(s * (1.0f / 64.0f));
}

// f32 -> f16, flat
__global__ void cvt_f16_kernel(const float* __restrict__ in, half_t* __restrict__ out, long n)
{
    long i = (long)blockIdx.x * blockDim.x + threadIdx.x;
    if (i < n) out[i] = (half_t)in[i];
}

// f32 -> f16 with row padding: out[r][c] = c<cols ? in[r][c] : 0
__global__ void cvt_f16_pad_kernel(const float* __restrict__ in, int ldi,
                                   half_t* __restrict__ out, int ldo,
                                   int rows, int cols)
{
    long i = (long)blockIdx.x * blockDim.x + threadIdx.x;
    long n = (long)rows * ldo;
    if (i >= n) return;
    int c = (int)(i % ldo);
    int r = (int)(i / ldo);
    out[i] = (c < cols) ? (half_t)in[(size_t)r * ldi + c] : (half_t)0.0f;
}

// ======================================================================
// Causal depthwise conv (width 4) + SiLU on xh half of xz.
// ======================================================================
__global__ __launch_bounds__(256)
void conv_silu_kernel(const float* __restrict__ xz,
                      const float* __restrict__ conv_w,
                      const float* __restrict__ conv_b,
                      float* __restrict__ xh32, half_t* __restrict__ xh16)
{
    long idx = (long)blockIdx.x * blockDim.x + threadIdx.x;
    if (idx >= (long)BL_ * DIN_) return;
    int d  = (int)(idx % DIN_);
    long bl = idx / DIN_;
    int l = (int)(bl % L_);
    long brow = bl - l;                      // b*L
    float acc = conv_b[d];
    #pragma unroll
    for (int k = 0; k < DCONV_; ++k) {
        int ls = l - (DCONV_ - 1) + k;
        if (ls >= 0)
            acc += xz[(brow + ls) * (size_t)(2 * DIN_) + d] * conv_w[d * DCONV_ + k];
    }
    float s = acc / (1.0f + __expf(-acc));   // SiLU
    xh32[idx] = s;
    xh16[idx] = (half_t)s;
}

// ======================================================================
// Fused selective scan. One thread per (b, d) channel; 16 states in regs.
// All 256 threads of a block share (b, l), so Bs/Cs (32 floats/step) are
// broadcast through double-buffered LDS: prefetch step l+1 while
// computing step l; one barrier per step.
// ======================================================================
__global__ __launch_bounds__(256)
void scan_kernel(const float* __restrict__ deltaraw,   // (BL x DIN)
                 const float* __restrict__ b_dt,       // (DIN)
                 const float* __restrict__ A_log,      // (DIN x DST)
                 const float* __restrict__ xh32,       // (BL x DIN)
                 const float* __restrict__ xdb,        // (BL x 80)
                 const float* __restrict__ D_param,    // (DIN)
                 const float* __restrict__ xz,         // (BL x 2*DIN)
                 half_t* __restrict__ y16)             // (BL x DIN)
{
    __shared__ float sBC[2][2 * DST_];                 // [buf][Bs(16) | Cs(16)]

    const int t = blockIdx.x * 256 + threadIdx.x;      // grid exact: 3072 threads
    const int d = t % DIN_;
    const int b = t / DIN_;                            // blocks never straddle b
    const size_t base = (size_t)b * L_;

    float A[DST_];
    #pragma unroll
    for (int s = 0; s < DST_; ++s) A[s] = -__expf(A_log[d * DST_ + s]);
    const float bd = b_dt[d];
    const float Dp = D_param[d];

    float h[DST_];
    #pragma unroll
    for (int s = 0; s < DST_; ++s) h[s] = 0.0f;

    if (threadIdx.x < 2 * DST_)
        sBC[0][threadIdx.x] = xdb[base * XDBW_ + DTR_ + threadIdx.x];
    __syncthreads();

    for (int l = 0; l < L_; ++l) {
        const size_t row = base + l;
        const int cur = l & 1;
        if ((l + 1 < L_) && threadIdx.x < 2 * DST_)    // prefetch next step's B/C
            sBC[cur ^ 1][threadIdx.x] = xdb[(row + 1) * XDBW_ + DTR_ + threadIdx.x];

        float draw = deltaraw[row * DIN_ + d] + bd;
        float delta = (draw > 20.0f) ? draw : __logf(1.0f + __expf(draw)); // softplus
        const float xt = xh32[row * DIN_ + d];
        const float zt = xz[row * (size_t)(2 * DIN_) + DIN_ + d];
        const float dx = delta * xt;
        float y = 0.0f;
        #pragma unroll
        for (int s = 0; s < DST_; ++s) {
            const float dA = __expf(delta * A[s]);
            h[s] = dA * h[s] + dx * sBC[cur][s];       // Bs broadcast
            y += h[s] * sBC[cur][DST_ + s];            // Cs broadcast
        }
        y += xt * Dp;
        y *= zt / (1.0f + __expf(-zt));                // * silu(z)
        y16[row * DIN_ + d] = (half_t)y;
        __syncthreads();
    }
}

// ======================================================================
// Host-side orchestration
// ======================================================================
static inline size_t align256(size_t v) { return (v + 255) & ~(size_t)255; }

extern "C" void kernel_launch(void* const* d_in, const int* in_sizes, int n_in,
                              void* d_out, int out_size, void* d_ws, size_t ws_size,
                              hipStream_t stream)
{
    (void)in_sizes; (void)n_in; (void)out_size; (void)ws_size;

    const float* x       = (const float*)d_in[0];
    const float* W_in    = (const float*)d_in[1];
    const float* conv_w  = (const float*)d_in[2];
    const float* conv_b  = (const float*)d_in[3];
    const float* W_xproj = (const float*)d_in[4];
    const float* W_dt    = (const float*)d_in[5];
    const float* b_dt    = (const float*)d_in[6];
    const float* A_log   = (const float*)d_in[7];
    const float* D_param = (const float*)d_in[8];
    const float* W_out   = (const float*)d_in[9];
    float* out = (float*)d_out;

    // ---- carve workspace ----
    char* ws = (char*)d_ws;
    size_t off = 0;
    auto carve = [&](size_t bytes) { char* p = ws + off; off += align256(bytes); return p; };

    half_t* w_in16  = (half_t*)carve((size_t)2 * DIN_ * C_ * sizeof(half_t));   // 3072x768
    half_t* xc16    = (half_t*)carve((size_t)BL_ * C_ * sizeof(half_t));        // 4096x768
    float*  xz      = (float*) carve((size_t)BL_ * 2 * DIN_ * sizeof(float));   // 4096x3072
    float*  xh32    = (float*) carve((size_t)BL_ * DIN_ * sizeof(float));       // 4096x1536
    half_t* xh16    = (half_t*)carve((size_t)BL_ * DIN_ * sizeof(half_t));
    half_t* w_xp16  = (half_t*)carve((size_t)XDBW_ * DIN_ * sizeof(half_t));    // 80x1536
    float*  xdb     = (float*) carve((size_t)BL_ * XDBW_ * sizeof(float));      // 4096x80
    half_t* dt16    = (half_t*)carve((size_t)BL_ * 64 * sizeof(half_t));        // 4096x64 (padded)
    half_t* w_dt16  = (half_t*)carve((size_t)DIN_ * 64 * sizeof(half_t));       // 1536x64 (padded)
    float*  draw    = (float*) carve((size_t)BL_ * DIN_ * sizeof(float));       // 4096x1536
    half_t* y16     = (half_t*)carve((size_t)BL_ * DIN_ * sizeof(half_t));
    half_t* w_out16 = (half_t*)carve((size_t)C_ * DIN_ * sizeof(half_t));       // 768x1536

    // ---- 1. weight conversions ----
    {
        long n = (long)2 * DIN_ * C_;
        cvt_f16_kernel<<<(n + 255) / 256, 256, 0, stream>>>(W_in, w_in16, n);
        n = (long)XDBW_ * DIN_;
        cvt_f16_kernel<<<(n + 255) / 256, 256, 0, stream>>>(W_xproj, w_xp16, n);
        n = (long)C_ * DIN_;
        cvt_f16_kernel<<<(n + 255) / 256, 256, 0, stream>>>(W_out, w_out16, n);
        long np = (long)DIN_ * 64;
        cvt_f16_pad_kernel<<<(np + 255) / 256, 256, 0, stream>>>(W_dt, DTR_, w_dt16, 64, DIN_, DTR_);
    }

    // ---- 2. xc = mean over H*W (f16) ----
    {
        long total = (long)BL_ * C_;                  // one wave each
        mean_hw_kernel<<<(unsigned)((total + 7) / 8), 256, 0, stream>>>(x, xc16, total);
    }

    // ---- 3. in-proj GEMM: xz = xc @ W_in^T   (M=4096,N=3072,K=768) ----
    launch_gemm(xc16, C_, w_in16, C_, xz, 2 * DIN_, BL_, 2 * DIN_, C_, stream);

    // ---- 4. depthwise causal conv + SiLU ----
    {
        long n = (long)BL_ * DIN_;
        conv_silu_kernel<<<(n + 255) / 256, 256, 0, stream>>>(xz, conv_w, conv_b, xh32, xh16);
    }

    // ---- 5. x-proj GEMM: xdb = xh @ W_xproj^T (M=4096,N=80,K=1536) ----
    launch_gemm(xh16, DIN_, w_xp16, DIN_, xdb, XDBW_, BL_, XDBW_, DIN_, stream);

    // ---- 6. pack dt (first 48 cols of xdb) into padded f16 (K=64) ----
    {
        long n = (long)BL_ * 64;
        cvt_f16_pad_kernel<<<(n + 255) / 256, 256, 0, stream>>>(xdb, XDBW_, dt16, 64, BL_, DTR_);
    }

    // ---- 7. dt-proj GEMM: draw = dt @ W_dt^T (M=4096,N=1536,K=64) ----
    launch_gemm(dt16, 64, w_dt16, 64, draw, DIN_, BL_, DIN_, 64, stream);

    // ---- 8. fused selective scan + D skip + SiLU(z) gate -> y16 ----
    scan_kernel<<<(B_ * DIN_) / 256, 256, 0, stream>>>(
        draw, b_dt, A_log, xh32, xdb, D_param, xz, y16);

    // ---- 9. out-proj GEMM: out = y @ W_out^T (M=4096,N=768,K=1536) ----
    launch_gemm(y16, DIN_, w_out16, DIN_, out, C_, BL_, C_, DIN_, stream);
}
